// ExoVariateEmbeddingMamba_68384469287334
// MI455X (gfx1250) — matrix operand order repaired and checked by reading, not compiled, not run on hardware
//
#include <hip/hip_runtime.h>
#include <hip/hip_bf16.h>

typedef __attribute__((ext_vector_type(16))) _Float16 v16h;
typedef __attribute__((ext_vector_type(8)))  float    v8f;

#define NSEQ 256   // B*V
#define LSEQ 1024
#define DM   128
#define SSTATE 64
#define RRANK 8
#define PCOLS 256  // B(64) | C(64) | dt_pre(128)
#define DOUT 512

// ---------------- Kernel A: fold small weights ----------------
// w_eff[j] = m_in_w[j]·in_w ; b_eff[j] = m_in_w[j]·in_b   (j<256)
// W_comb rows 0..127  = xproj_w[8..135]  (B then C, contiguous)
// W_comb rows 128..255 = dt_w @ xproj_w[:8]
// Aneg = -exp(A_log)
__global__ void precompute_kernel(const float* __restrict__ in_w, const float* __restrict__ in_b,
                                  const float* __restrict__ m_in_w, const float* __restrict__ xproj_w,
                                  const float* __restrict__ dt_w, const float* __restrict__ A_log,
                                  float* __restrict__ w_eff, float* __restrict__ b_eff,
                                  float* __restrict__ Aneg, _Float16* __restrict__ Wc) {
  const int t = threadIdx.x; // 256 threads
  float sw = 0.f, sb = 0.f;
  for (int k = 0; k < DM; ++k) {
    float mw = m_in_w[(size_t)t * DM + k];
    sw += mw * in_w[k];
    sb += mw * in_b[k];
  }
  w_eff[t] = sw;
  b_eff[t] = sb;

  if (t < 128) {
    const float* src = xproj_w + (size_t)(RRANK + t) * DM;
    for (int k = 0; k < DM; ++k) Wc[(size_t)t * DM + k] = (_Float16)src[k];
  } else {
    const int c = t - 128;
    for (int k = 0; k < DM; ++k) {
      float s = 0.f;
      for (int q = 0; q < RRANK; ++q) s += dt_w[c * RRANK + q] * xproj_w[(size_t)q * DM + k];
      Wc[(size_t)t * DM + k] = (_Float16)s;
    }
  }
  for (int i = t; i < DM * SSTATE; i += 256) Aneg[i] = -__expf(A_log[i]);
}

// W_final[o,k] = sum_j out_w[o,j] * m_out_w[j,k]   (512 x 128)
__global__ void precompute2_kernel(const float* __restrict__ out_w, const float* __restrict__ m_out_w,
                                   float* __restrict__ W_final) {
  const int o = blockIdx.x;   // 512
  const int k = threadIdx.x;  // 128
  float s = 0.f;
  for (int j = 0; j < DM; ++j) s += out_w[(size_t)o * DM + j] * m_out_w[(size_t)j * DM + k];
  W_final[(size_t)o * DM + k] = s;
}

// ---------------- Kernel B: fused embed + causal conv(kc=2) + SiLU ----------------
__global__ void embed_conv_kernel(const float* __restrict__ x_exo, const float* __restrict__ w_eff,
                                  const float* __restrict__ b_eff, const float* __restrict__ conv_w,
                                  const float* __restrict__ conv_b, float* __restrict__ xcs) {
  const int l = blockIdx.x, n = blockIdx.y, c = threadIdx.x; // 128 threads
  const int b = n >> 6, v = n & 63;
  const float we = w_eff[c], be = b_eff[c];
  const float sx = x_exo[((size_t)b * LSEQ + l) * 64 + v];
  const float x1 = sx * we + be;
  float x0 = 0.f;
  if (l > 0) {
    const float sxp = x_exo[((size_t)b * LSEQ + (l - 1)) * 64 + v];
    x0 = sxp * we + be;
  }
  float acc = conv_w[c * 2 + 0] * x0 + conv_w[c * 2 + 1] * x1 + conv_b[c];
  const float sig = 1.f / (1.f + __expf(-acc));
  xcs[((size_t)n * LSEQ + l) * DM + c] = acc * sig;
}

// ---------------- Kernel C: WMMA projection GEMM ----------------
// P(262144 x 256) = xcs(262144 x 128) @ W_comb^T.
// One block per 128-row panel; full W_comb (64KB f16) async-staged into LDS
// ([col][k] layout == its global layout -> straight async memcpy).
// Each wave keeps its 16x128 A panel in registers (4 x v16h) and sweeps all
// 16 column tiles -> 64 WMMAs/wave; B operand = contiguous 32B DS loads.
__global__ void gemm_proj_kernel(const float* __restrict__ xc, const _Float16* __restrict__ Wc,
                                 float* __restrict__ P) {
  __shared__ __align__(64) _Float16 Bsh[PCOLS * DM]; // 64 KB, [col][k]
  const int tid = threadIdx.x; // 256

  // Async global->LDS stage of W_comb: 16B per lane per issue, ASYNCcnt-tracked.
  {
    const uint32_t lds0 = (uint32_t)(uintptr_t)(&Bsh[0]);
    const uint64_t g0 = (uint64_t)(uintptr_t)Wc;
    for (int i = tid * 16; i < PCOLS * DM * 2; i += 256 * 16) {
      uint32_t lds = lds0 + (uint32_t)i;
      uint64_t g = g0 + (uint64_t)i;
      asm volatile("global_load_async_to_lds_b128 %0, %1, off"
                   :: "v"(lds), "v"(g) : "memory");
    }
    asm volatile("s_wait_asynccnt 0" ::: "memory");
  }
  __syncthreads();

  const int wave = tid >> 5;
  const int lane = tid & 31;
  const int laneHi = lane >> 4; // wave32 half
  const int m = lane & 15;
  const size_t r0 = (size_t)blockIdx.x * 128 + (size_t)wave * 16;
  const float* Arow = xc + (r0 + m) * DM;
  __builtin_prefetch(Arow, 0, 1); // -> global_prefetch_b8

  // A panel in registers: element e -> K = kt*32 + laneHi*16 + e (matches B order):
  // lanes 0-15 K=0..15, lanes 16-31 K=16..31.
  v16h a[4];
#pragma unroll
  for (int kt = 0; kt < 4; ++kt) {
    const int K0 = kt * 32;
#pragma unroll
    for (int e = 0; e < 16; ++e) {
      const int vg = e >> 1, p = e & 1;
      const int base = (vg < 4) ? (vg * 2) : (16 + (vg - 4) * 2);
      a[kt][e] = (_Float16)Arow[K0 + laneHi * 16 + e];
    }
  }

  float* Pw = P + r0 * PCOLS + (laneHi ? 8 * PCOLS : 0) + m;
  for (int ct = 0; ct < 16; ++ct) {
    const int c0 = ct * 16;
    v8f acc = {};
#pragma unroll
    for (int kt = 0; kt < 4; ++kt) {
      // B 32x16: element e -> K = kt*32 + laneHi*16 + e, col = lane%16 (contiguous in LDS)
      const v16h b = *(const v16h*)&Bsh[(size_t)(c0 + m) * DM + kt * 32 + laneHi * 16];
      acc = __builtin_amdgcn_wmma_f32_16x16x32_f16(false, a[kt], false, b, (short)0, acc,
                                                   false, false);
    }
#pragma unroll
    for (int vv = 0; vv < 8; ++vv) Pw[(size_t)vv * PCOLS + c0] = acc[vv];
  }
}

// ---------------- Kernel D: selective-scan (sequential over L) ----------------
// 1 block / sequence; thread t: channel c=t>>1, state-half h=t&1 (32 states in regs).
__global__ void scan_kernel(const float* __restrict__ x_exo, const float* __restrict__ P,
                            const float* __restrict__ xcs, const float* __restrict__ Aneg,
                            const float* __restrict__ dt_b, const float* __restrict__ Dv,
                            const float* __restrict__ w_eff, const float* __restrict__ b_eff,
                            float* __restrict__ Y) {
  __shared__ float BC[132]; // [0..63]=B_t, [64..127]=C_t, [128]=sx
  const int n = blockIdx.x, t = threadIdx.x; // 256
  const int c = t >> 1, h = t & 1;
  const int b = n >> 6, v = n & 63;

  float hreg[32], Arow[32];
#pragma unroll
  for (int j = 0; j < 32; ++j) {
    hreg[j] = 0.f;
    Arow[j] = Aneg[c * SSTATE + h * 32 + j];
  }
  const float dtb = dt_b[c], Dc = Dv[c];
  const float wz = w_eff[128 + c], bz = b_eff[128 + c];
  float ysum = 0.f;
  const float* Pn = P + (size_t)n * LSEQ * PCOLS;
  const float* Xn = xcs + (size_t)n * LSEQ * DM;

  for (int l = 0; l < LSEQ; ++l) {
    __syncthreads();
    if (t < 128) BC[t] = Pn[(size_t)l * PCOLS + t];
    if (t == 128) BC[128] = x_exo[((size_t)b * LSEQ + l) * 64 + v];
    __syncthreads();

    const float dtp = Pn[(size_t)l * PCOLS + 128 + c];
    const float zp = dtp + dtb;
    const float dt = (zp > 20.f) ? zp : log1pf(__expf(zp)); // softplus
    const float xcv = Xn[(size_t)l * DM + c];
    const float dtx = dt * xcv;
    const float* Bp = BC + h * 32;
    const float* Cp = BC + 64 + h * 32;
    float yp = 0.f;
#pragma unroll
    for (int j = 0; j < 32; ++j) {
      const float dA = __expf(dt * Arow[j]);
      hreg[j] = hreg[j] * dA + dtx * Bp[j];
      yp += hreg[j] * Cp[j];
    }
    yp += __shfl_xor(yp, 1, 32); // partner lane (same wave32) holds other 32 states
    if (h == 0) {
      const float sx = BC[128];
      const float zz = sx * wz + bz;
      const float sz = zz / (1.f + __expf(-zz)); // silu(z)
      ysum += (yp + Dc * xcv) * sz;
    }
  }
  if (h == 0) Y[(size_t)n * DM + c] = ysum * (1.f / (float)LSEQ); // pooled mean
}

// ---------------- Kernel E: out = pooled @ W_final^T + out_b ----------------
__global__ void out_kernel(const float* __restrict__ Y, const float* __restrict__ W_final,
                           const float* __restrict__ out_b, float* __restrict__ out) {
  __shared__ float ysh[DM];
  const int n = blockIdx.x, half = blockIdx.y;
  const int tl = threadIdx.x; // 256
  if (tl < DM) ysh[tl] = Y[(size_t)n * DM + tl];
  __syncthreads();
  const int o = half * 256 + tl;
  float s = 0.f;
  for (int k = 0; k < DM; ++k) s += ysh[k] * W_final[(size_t)o * DM + k];
  out[(size_t)n * DOUT + o] = s + out_b[o];
}

extern "C" void kernel_launch(void* const* d_in, const int* in_sizes, int n_in,
                              void* d_out, int out_size, void* d_ws, size_t ws_size,
                              hipStream_t stream) {
  const float* x_exo  = (const float*)d_in[0];
  const float* in_w   = (const float*)d_in[1];
  const float* in_b   = (const float*)d_in[2];
  const float* m_in_w = (const float*)d_in[3];
  const float* conv_w = (const float*)d_in[4];
  const float* conv_b = (const float*)d_in[5];
  const float* xproj_w= (const float*)d_in[6];
  const float* dt_w   = (const float*)d_in[7];
  const float* dt_b   = (const float*)d_in[8];
  const float* A_log  = (const float*)d_in[9];
  const float* Dvec   = (const float*)d_in[10];
  const float* m_out_w= (const float*)d_in[11];
  const float* out_w  = (const float*)d_in[12];
  const float* out_b  = (const float*)d_in[13];
  float* outp = (float*)d_out;

  uint8_t* ws = (uint8_t*)d_ws;
  float*    w_eff   = (float*)   (ws + 0);          //   1 KB
  float*    b_eff   = (float*)   (ws + 1024);       //   1 KB
  float*    Aneg    = (float*)   (ws + 2048);       //  32 KB
  _Float16* Wc      = (_Float16*)(ws + 34816);      //  64 KB
  float*    W_final = (float*)   (ws + 100352);     // 256 KB
  float*    Yp      = (float*)   (ws + 362496);     // 128 KB
  float*    xcs     = (float*)   (ws + 493568);     // 128 MB
  float*    Pm      = (float*)   (ws + 134711296);  // 256 MB

  precompute_kernel<<<1, 256, 0, stream>>>(in_w, in_b, m_in_w, xproj_w, dt_w, A_log,
                                           w_eff, b_eff, Aneg, Wc);
  precompute2_kernel<<<DOUT, DM, 0, stream>>>(out_w, m_out_w, W_final);
  embed_conv_kernel<<<dim3(LSEQ, NSEQ), DM, 0, stream>>>(x_exo, w_eff, b_eff, conv_w, conv_b, xcs);
  gemm_proj_kernel<<<dim3((NSEQ * LSEQ) / 128, 1), 256, 0, stream>>>(xcs, Wc, Pm);
  scan_kernel<<<NSEQ, 256, 0, stream>>>(x_exo, Pm, xcs, Aneg, dt_b, Dvec, w_eff, b_eff, Yp);
  out_kernel<<<dim3(NSEQ, 2), 256, 0, stream>>>(Yp, W_final, out_b, outp);
}
